// LightGCN_1228360647043
// MI455X (gfx1250) — matrix operand order repaired and checked by reading, not compiled, not run on hardware
//
#include <hip/hip_runtime.h>
#include <hip/hip_bf16.h>

#define NUM_USERS 60000
#define NUM_ITEMS 40000
#define NUM_NODES (NUM_USERS + NUM_ITEMS)
#define EMBED_DIM 64
#define NUM_LAYERS 3
#define NUM_EDGES 1200000
#define BATCH 16384

typedef __attribute__((ext_vector_type(16))) _Float16 v16h;
typedef __attribute__((ext_vector_type(8)))  float    v8f;

// Hardware f32 atomic add at device scope (L2 RMW), no-return -> STOREcnt.
__device__ __forceinline__ void atomic_add_f32_dev(float* p, float v) {
    asm volatile("global_atomic_add_f32 %0, %1, off scope:SCOPE_DEV"
                 :: "v"(p), "v"(v) : "memory");
}

// ---------------- concat user_emb | item_emb -> X (float4 granularity) -----
#define TOTAL_F4  ((size_t)NUM_NODES * EMBED_DIM / 4)   // 1,600,000
#define USER_F4   ((size_t)NUM_USERS * EMBED_DIM / 4)   //   960,000

__global__ void concat_kernel(const float4* __restrict__ ue,
                              const float4* __restrict__ ie,
                              float4* __restrict__ X) {
    size_t i = (size_t)blockIdx.x * blockDim.x + threadIdx.x;
    if (i < USER_F4)        X[i] = ue[i];
    else if (i < TOTAL_F4)  X[i] = ie[i - USER_F4];
}

__global__ void zero_kernel(float4* __restrict__ Y) {
    size_t i = (size_t)blockIdx.x * blockDim.x + threadIdx.x;
    if (i < TOTAL_F4) Y[i] = make_float4(0.f, 0.f, 0.f, 0.f);
}

// ---------------- SpMM: one wave32 per edge, float2 per lane ---------------
__global__ void spmm_kernel(const int*   __restrict__ rows,
                            const int*   __restrict__ cols,
                            const float* __restrict__ vals,
                            const float* __restrict__ x,
                            float*       __restrict__ y) {
    const size_t t = (size_t)blockIdx.x * blockDim.x + threadIdx.x;
    const int e    = (int)(t >> 5);
    const int lane = (int)(t & 31);
    if (e >= NUM_EDGES) return;

    const int   r = rows[e];
    const int   c = cols[e];
    const float v = vals[e];

    const float2 xv = *(const float2*)(x + (size_t)c * EMBED_DIM + lane * 2);
    float* p = y + (size_t)r * EMBED_DIM + lane * 2;
    atomic_add_f32_dev(p + 0, v * xv.x);
    atomic_add_f32_dev(p + 1, v * xv.y);
}

// ---------------- batched dot products via WMMA: diag(U x V^T) -------------
// One wave handles 16 batch elements. A = U tile (16x32 f16), B = V^T tile
// (32x16 f16), C accumulates f32; K=64 done as two K=32 WMMAs.
__global__ void dot_wmma_kernel(const float* __restrict__ nodes,
                                const int*   __restrict__ uidx,
                                const int*   __restrict__ iidx,
                                float*       __restrict__ out) {
    const int lane = threadIdx.x & 31;
    const int wave = (int)(((size_t)blockIdx.x * blockDim.x + threadIdx.x) >> 5);
    const int b0   = wave * 16;                  // 16 pairs per wave
    const int m    = lane & 15;                  // row/col index within tile
    const int hi   = lane >> 4;                  // lane half (K split)

    const int urow = uidx[b0 + m];
    const int irow = iidx[b0 + m] + NUM_USERS;
    const float* U = nodes + (size_t)urow * EMBED_DIM;
    const float* V = nodes + (size_t)irow * EMBED_DIM;

    v8f c = {};
#pragma unroll
    for (int kc = 0; kc < 2; ++kc) {
        const int kb = kc * 32;
        v16h a, b;
        // A (16x32 f16): lane holds row m, 16 consecutive K values
        // (lanes 0..15 -> K 0..15, lanes 16..31 -> K 16..31 of the chunk).
#pragma unroll
        for (int i = 0; i < 16; ++i)
            a[i] = (_Float16)U[kb + hi * 16 + i];
        // B (32x16 f16): lane holds column m, 16 consecutive K values
        // (lanes 0..15 -> K 0..15, lanes 16..31 -> K 16..31 of the chunk).
#pragma unroll
        for (int i = 0; i < 16; ++i)
            b[i] = (_Float16)V[kb + hi * 16 + i];

        c = __builtin_amdgcn_wmma_f32_16x16x32_f16(
                /*neg_a=*/false, a, /*neg_b=*/false, b,
                /*c_mod=*/(short)0, c, /*reuse_a=*/false, /*reuse_b=*/false);
    }

    // Diagonal of C (16x16 f32, 8 VGPRs):
    //   m in 0..7  -> lane m,     element c[m]
    //   m in 8..15 -> lane m+16,  element c[m-8]
    if (lane < 8)        out[b0 + lane]        = c[lane];
    else if (lane >= 24) out[b0 + (lane - 16)] = c[lane - 24];
}

extern "C" void kernel_launch(void* const* d_in, const int* in_sizes, int n_in,
                              void* d_out, int out_size, void* d_ws, size_t ws_size,
                              hipStream_t stream) {
    const int*   uidx  = (const int*)  d_in[0];
    const int*   iidx  = (const int*)  d_in[1];
    const int*   erows = (const int*)  d_in[2];
    const int*   ecols = (const int*)  d_in[3];
    const float* evals = (const float*)d_in[4];
    const float* uemb  = (const float*)d_in[5];
    const float* iemb  = (const float*)d_in[6];
    float* out = (float*)d_out;

    // Ping-pong node-feature buffers in workspace (2 x 25.6 MB).
    float* X = (float*)d_ws;
    float* Y = X + (size_t)NUM_NODES * EMBED_DIM;

    const int elem_blocks = (int)((TOTAL_F4 + 255) / 256);          // 6250
    concat_kernel<<<elem_blocks, 256, 0, stream>>>(
        (const float4*)uemb, (const float4*)iemb, (float4*)X);

    const int spmm_blocks = (int)(((size_t)NUM_EDGES * 32 + 255) / 256); // 150000
    for (int layer = 0; layer < NUM_LAYERS; ++layer) {
        zero_kernel<<<elem_blocks, 256, 0, stream>>>((float4*)Y);
        spmm_kernel<<<spmm_blocks, 256, 0, stream>>>(erows, ecols, evals, X, Y);
        float* t = X; X = Y; Y = t;   // result of this layer now in X
    }

    // 16384 pairs, 16 per wave, 8 waves per block -> 128 blocks.
    dot_wmma_kernel<<<BATCH / 128, 256, 0, stream>>>(X, uidx, iidx, out);
}
